// NKF_29240137351429
// MI455X (gfx1250) — matrix operand order repaired and checked by reading, not compiled, not run on hardware
//
#include <hip/hip_runtime.h>
#include <hip/hip_bf16.h>
#include <stdint.h>
#include <math.h>

// ---- WMMA vector types (gfx1250, wave32) ----
typedef __attribute__((ext_vector_type(16))) _Float16 v16h;
typedef __attribute__((ext_vector_type(8)))  float    v8f;

// ---- problem constants ----
#define B_    4
#define N_    25600
#define T_    101
#define TP2_  104            // T + 3 leading zero columns (window history)
#define F_    513
#define BF_   2052           // B*F channels
#define FC_   68
#define RD_   68
#define NG_   204            // 3*RD
#define NGP_  208            // padded N for GRU gemm
#define KT_   3              // K tiles of 32 (68 -> 96)
#define MT_   129            // ceil(2052/16)
#define MP_   2064
#define NTG_  13             // 204 -> 208 (13 tiles of 16)
#define NT2_  5              // 68 -> 80
#define NP2_  80
#define MRES_ 25
#define NFFT_ 1024
#define HOP_  256

#define FRAGH_   (MT_*KT_*32*16)     // halves in one A-fragment array (full M)
#define WFRAGH_  (NTG_*KT_*32*16)    // halves in one GRU B-fragment array
#define W2FRAGH_ (NT2_*KT_*32*16)    // halves in one dense2 B-fragment array
#define GSTRIDE_  ((size_t)MP_*NGP_)
#define G2STRIDE_ ((size_t)MP_*NP2_)

// A-fragment (16x32 f16) element placement per CDNA5 ISA layout table.
__device__ __forceinline__ void afrag_store(_Float16* frag, int m, int k, float v) {
  int mt = m >> 4, kt = k >> 5, kl = k & 31;
  int hi = (kl >> 3) & 1;                       // lane-half selector
  int j  = (kl & 7) + ((kl >> 4) << 3);         // half index within lane
  int lane = (m & 15) + (hi << 4);
  frag[((size_t)((mt*KT_ + kt)*32 + lane))*16 + j] = (_Float16)v;
}

__device__ __forceinline__ float sigmf_(float x) { return 1.0f/(1.0f + expf(-x)); }

// ---------------- zero init (runs every call: deterministic state) ----------------
__global__ void k_zero(uint32_t* p, size_t n) {
  size_t i = (size_t)blockIdx.x*blockDim.x + threadIdx.x;
  size_t st = (size_t)gridDim.x*blockDim.x;
  for (; i < n; i += st) p[i] = 0u;
}

// ---------------- weight -> B-fragment swizzle (one-time) ----------------
// B tile is 32x16 (K x N); lane L holds column n=L&15, 16 halves cover 32 k's.
__global__ void k_prep_bfrag(const float* __restrict__ W, _Float16* __restrict__ dst,
                             int Nrows, int Kc, int ntiles_n) {
  int e = blockIdx.x*blockDim.x + threadIdx.x;
  int total = ntiles_n*KT_*512;
  if (e >= total) return;
  int tile = e >> 9, rem = e & 511, lane = rem >> 4, h = rem & 15;
  int nt = tile / KT_, kt = tile % KT_;
  int n = nt*16 + (lane & 15);
  int k = kt*32 + (h & 7) + ((lane >> 4) << 3) + ((h >> 3) << 4);
  dst[e] = (n < Nrows && k < Kc) ? (_Float16)W[(size_t)n*Kc + k] : (_Float16)0.0f;
}

// ---------------- STFT: hann window + 1024-pt radix-2 FFT per frame ----------------
__global__ void k_stft(const float* __restrict__ x, const float* __restrict__ y,
                       float* __restrict__ Xfp, float* __restrict__ Yf) {
  __shared__ float re[NFFT_], im[NFFT_];
  int idx = blockIdx.x;
  int which = idx / (B_*T_);
  int bt = idx % (B_*T_);
  int b = bt / T_, t = bt % T_;
  const float* sig = which ? y : x;
  int tid = threadIdx.x;
  for (int n = tid; n < NFFT_; n += blockDim.x) {
    int j = t*HOP_ + n - NFFT_/2;            // center=True reflect pad
    if (j < 0) j = -j;
    if (j >= N_) j = 2*N_ - 2 - j;
    float w = 0.5f*(1.0f - cosf(6.28318530717958647692f * (float)n / (float)NFFT_));
    re[n] = sig[(size_t)b*N_ + j] * w;
    im[n] = 0.0f;
  }
  __syncthreads();
  for (int n = tid; n < NFFT_; n += blockDim.x) {
    int r = (int)(__brev((unsigned)n) >> 22);
    if (r > n) { float a=re[n]; re[n]=re[r]; re[r]=a; float c=im[n]; im[n]=im[r]; im[r]=c; }
  }
  __syncthreads();
  for (int s = 1; s <= 10; ++s) {
    int half = 1 << (s-1);
    for (int i = tid; i < NFFT_/2; i += blockDim.x) {
      int grp = i >> (s-1), pos = i & (half-1);
      int i0 = (grp << s) + pos, i1 = i0 + half;
      float ang = -6.28318530717958647692f * (float)pos / (float)(1 << s);
      float wr = cosf(ang), wi = sinf(ang);
      float tr = wr*re[i1] - wi*im[i1];
      float ti = wr*im[i1] + wi*re[i1];
      re[i1] = re[i0] - tr; im[i1] = im[i0] - ti;
      re[i0] += tr;         im[i0] += ti;
    }
    __syncthreads();
  }
  for (int f = tid; f < F_; f += blockDim.x) {
    size_t c = (size_t)b*F_ + f;
    if (which == 0) {
      Xfp[(c*TP2_ + (t+3))*2 + 0] = re[f];
      Xfp[(c*TP2_ + (t+3))*2 + 1] = im[f];
    } else {
      Yf[(c*T_ + t)*2 + 0] = re[f];
      Yf[(c*T_ + t)*2 + 1] = im[f];
    }
  }
}

// ---------------- per-step: innovation + input dense -> z fragments ----------------
__global__ void k_pre(const float* __restrict__ Xfp, const float* __restrict__ Yf,
                      float* __restrict__ h_prior, const float* __restrict__ h_post,
                      float* __restrict__ e_buf,
                      _Float16* __restrict__ zfragR, _Float16* __restrict__ zfragI,
                      const float* __restrict__ Wr1, const float* __restrict__ Wi1,
                      const float* __restrict__ br1, const float* __restrict__ bi1,
                      const float* __restrict__ a_in, int t) {
  int c = blockIdx.x*blockDim.x + threadIdx.x;
  if (c >= BF_) return;
  float hq[8], hp[8];
  #pragma unroll
  for (int i=0;i<8;++i){ hq[i]=h_post[(size_t)c*8+i]; hp[i]=h_prior[(size_t)c*8+i]; }
  const float* xr = Xfp + ((size_t)c*TP2_ + t)*2;
  float xtr[4], xti[4];
  #pragma unroll
  for (int l=0;l<4;++l){ xtr[l]=xr[2*l]; xti[l]=xr[2*l+1]; }
  float ytr = Yf[((size_t)c*T_ + t)*2 + 0];
  float yti = Yf[((size_t)c*T_ + t)*2 + 1];
  float er = ytr, ei = yti;
  #pragma unroll
  for (int l=0;l<4;++l){
    er -= xtr[l]*hq[2*l]   - xti[l]*hq[2*l+1];
    ei -= xtr[l]*hq[2*l+1] + xti[l]*hq[2*l];
  }
  #pragma unroll
  for (int i=0;i<8;++i) h_prior[(size_t)c*8+i] = hq[i];   // h_prior <- h_post
  e_buf[(size_t)c*2+0]=er; e_buf[(size_t)c*2+1]=ei;
  float fr[9], fi[9];
  #pragma unroll
  for (int l=0;l<4;++l){ fr[l]=xtr[l]; fi[l]=xti[l]; }
  fr[4]=er; fi[4]=ei;
  #pragma unroll
  for (int l=0;l<4;++l){ fr[5+l]=hq[2*l]-hp[2*l]; fi[5+l]=hq[2*l+1]-hp[2*l+1]; }
  float a = a_in[0];
  for (int j=0;j<FC_;++j){
    float sr=br1[j], si=bi1[j];
    #pragma unroll
    for (int k=0;k<9;++k){ sr += fr[k]*Wr1[j*9+k]; si += fi[k]*Wi1[j*9+k]; }
    sr = sr>=0.f ? sr : a*sr;
    si = si>=0.f ? si : a*si;
    afrag_store(zfragR, c, j, sr);
    afrag_store(zfragI, c, j, si);
  }
}

// ---------------- per-step: 8 GRU GEMMs via WMMA f16 -> f32 ----------------
__global__ void k_gru_gemm(const _Float16* __restrict__ zfragR, const _Float16* __restrict__ zfragI,
                           const _Float16* __restrict__ hfrag,
                           const _Float16* __restrict__ wih, const _Float16* __restrict__ whh,
                           float* __restrict__ Gi, float* __restrict__ Gh) {
  int w = (blockIdx.x*blockDim.x + threadIdx.x) >> 5;
  int lane = threadIdx.x & 31;
  const int TILES = 4*MT_*NTG_;
  if (w >= TILES) return;            // wave-uniform guard (EXEC all-ones for WMMA)
  int e  = w / (MT_*NTG_);
  int r  = w % (MT_*NTG_);
  int mt = r / NTG_, nt = r % NTG_;
  const _Float16* az = (e & 1) ? zfragI : zfragR;
  const _Float16* ah = hfrag + (size_t)e * FRAGH_;
  const _Float16* bi = wih + (size_t)(e >> 1) * WFRAGH_;
  const _Float16* bh = whh + (size_t)(e >> 1) * WFRAGH_;
  v8f ci = {}; v8f ch = {};
  #pragma unroll
  for (int kt=0; kt<KT_; ++kt) {
    v16h a0 = *(const v16h*)(az + ((size_t)((mt*KT_+kt)*32 + lane))*16);
    v16h a1 = *(const v16h*)(ah + ((size_t)((mt*KT_+kt)*32 + lane))*16);
    v16h b0 = *(const v16h*)(bi + ((size_t)((nt*KT_+kt)*32 + lane))*16);
    v16h b1 = *(const v16h*)(bh + ((size_t)((nt*KT_+kt)*32 + lane))*16);
    ci = __builtin_amdgcn_wmma_f32_16x16x32_f16(false, a0, false, b0, (short)0, ci, false, false);
    ch = __builtin_amdgcn_wmma_f32_16x16x32_f16(false, a1, false, b1, (short)0, ch, false, false);
  }
  float* gi = Gi + (size_t)e * GSTRIDE_;
  float* gh = Gh + (size_t)e * GSTRIDE_;
  int n    = nt*16 + (lane & 15);
  int mrow = mt*16 + ((lane >> 4) << 3);
  #pragma unroll
  for (int j=0;j<8;++j){
    gi[(size_t)(mrow+j)*NGP_ + n] = ci[j];
    gh[(size_t)(mrow+j)*NGP_ + n] = ch[j];
  }
}

// ---------------- per-step: GRU gate combine + state update ----------------
__global__ void k_gru_combine(const float* __restrict__ Gi, const float* __restrict__ Gh,
                              const float* __restrict__ bih_r, const float* __restrict__ bhh_r,
                              const float* __restrict__ bih_i, const float* __restrict__ bhh_i,
                              float* __restrict__ h32, _Float16* __restrict__ hfrag,
                              _Float16* __restrict__ ufrag, _Float16* __restrict__ vfrag) {
  int idx = blockIdx.x*blockDim.x + threadIdx.x;
  if (idx >= BF_*RD_) return;
  int c = idx / RD_, u = idx % RD_;
  float hn[4];
  #pragma unroll
  for (int e=0;e<4;++e){
    const float* gi = Gi + (size_t)e*GSTRIDE_ + (size_t)c*NGP_;
    const float* gh = Gh + (size_t)e*GSTRIDE_ + (size_t)c*NGP_;
    const float* bi = (e<2)? bih_r : bih_i;
    const float* bh = (e<2)? bhh_r : bhh_i;
    float rg  = sigmf_(gi[u]        + bi[u]        + gh[u]        + bh[u]);
    float zg  = sigmf_(gi[RD_+u]    + bi[RD_+u]    + gh[RD_+u]    + bh[RD_+u]);
    float ng  = tanhf (gi[2*RD_+u]  + bi[2*RD_+u]  + rg*(gh[2*RD_+u] + bh[2*RD_+u]));
    float hold = h32[(size_t)e*BF_*RD_ + idx];
    float hv = (1.f - zg)*ng + zg*hold;
    h32[(size_t)e*BF_*RD_ + idx] = hv;
    afrag_store(hfrag + (size_t)e*FRAGH_, c, u, hv);
    hn[e] = hv;
  }
  afrag_store(ufrag, c, u, hn[0] - hn[3]);   // Re(rnn_out) = h_rr - h_ii
  afrag_store(vfrag, c, u, hn[2] + hn[1]);   // Im(rnn_out) = h_ri + h_ir
}

// ---------------- per-step: complex dense2 (68->68) via WMMA ----------------
__global__ void k_dense2_gemm(const _Float16* __restrict__ ufrag, const _Float16* __restrict__ vfrag,
                              const _Float16* __restrict__ w2frag, float* __restrict__ G2) {
  int w = (blockIdx.x*blockDim.x + threadIdx.x) >> 5;
  int lane = threadIdx.x & 31;
  const int TILES = 2*MT_*NT2_;
  if (w >= TILES) return;
  int mat = w / (MT_*NT2_);
  int r   = w % (MT_*NT2_);
  int mt = r / NT2_, nt = r % NT2_;
  const _Float16* af = mat ? vfrag : ufrag;
  const _Float16* bf = w2frag + (size_t)mat * W2FRAGH_;
  v8f cacc = {};
  #pragma unroll
  for (int kt=0; kt<KT_; ++kt) {
    v16h a0 = *(const v16h*)(af + ((size_t)((mt*KT_+kt)*32 + lane))*16);
    v16h b0 = *(const v16h*)(bf + ((size_t)((nt*KT_+kt)*32 + lane))*16);
    cacc = __builtin_amdgcn_wmma_f32_16x16x32_f16(false, a0, false, b0, (short)0, cacc, false, false);
  }
  float* g = G2 + (size_t)mat * G2STRIDE_;
  int n    = nt*16 + (lane & 15);
  int mrow = mt*16 + ((lane >> 4) << 3);
  #pragma unroll
  for (int j=0;j<8;++j) g[(size_t)(mrow+j)*NP2_ + n] = cacc[j];
}

// ---------------- per-step: PReLU + dense3 + Kalman update + echo + resf ----------------
__global__ void k_kalman(const float* __restrict__ G2,
                         const float* __restrict__ br2, const float* __restrict__ bi2,
                         const float* __restrict__ a_out,
                         const float* __restrict__ Wr3, const float* __restrict__ Wi3,
                         const float* __restrict__ br3, const float* __restrict__ bi3,
                         const float* __restrict__ Xfp, const float* __restrict__ Yf,
                         const float* __restrict__ e_buf, const float* __restrict__ h_prior,
                         float* __restrict__ h_post, float* __restrict__ echo,
                         float* __restrict__ resf, int t) {
  int c = blockIdx.x*blockDim.x + threadIdx.x;
  if (c >= BF_) return;
  float a = a_out[0];
  const float* g2r = G2 + (size_t)c*NP2_;
  const float* g2i = G2 + G2STRIDE_ + (size_t)c*NP2_;
  float kr[4], ki[4];
  #pragma unroll
  for (int l=0;l<4;++l){ kr[l]=br3[l]; ki[l]=bi3[l]; }
  for (int j=0;j<FC_;++j){
    float zr = g2r[j] + br2[j]; zr = zr>=0.f ? zr : a*zr;
    float zi = g2i[j] + bi2[j]; zi = zi>=0.f ? zi : a*zi;
    #pragma unroll
    for (int l=0;l<4;++l){ kr[l]+=zr*Wr3[l*FC_+j]; ki[l]+=zi*Wi3[l*FC_+j]; }
  }
  float er=e_buf[(size_t)c*2], ei=e_buf[(size_t)c*2+1];
  float hp[8];
  #pragma unroll
  for (int i=0;i<8;++i) hp[i]=h_prior[(size_t)c*8+i];
  float pr[4], pi[4];
  #pragma unroll
  for (int l=0;l<4;++l){
    float dr = kr[l]*er - ki[l]*ei;
    float di = kr[l]*ei + ki[l]*er;
    float vr = fminf(fmaxf(hp[2*l]  +dr,-8.f),8.f);
    float vi = fminf(fmaxf(hp[2*l+1]+di,-8.f),8.f);
    pr[l]=vr; pi[l]=vi;
    h_post[(size_t)c*8+2*l]=vr; h_post[(size_t)c*8+2*l+1]=vi;
  }
  const float* xr = Xfp + ((size_t)c*TP2_ + t)*2;
  float exr=0.f, exi=0.f;
  #pragma unroll
  for (int l=0;l<4;++l){
    float xtr=xr[2*l], xti=xr[2*l+1];
    exr += xtr*pr[l] - xti*pi[l];
    exi += xtr*pi[l] + xti*pr[l];
  }
  exr = fminf(fmaxf(exr,-10.f),10.f);
  exi = fminf(fmaxf(exi,-10.f),10.f);
  echo[(size_t)c*2]=exr; echo[(size_t)c*2+1]=exi;
  float ytr=Yf[((size_t)c*T_+t)*2], yti=Yf[((size_t)c*T_+t)*2+1];
  resf[c] = ytr*ytr + yti*yti - exr*exr - exi*exi;
}

// ---------------- per-step: residual GRU path + mask + write outputs ----------------
__global__ void k_res(const float* __restrict__ resf,
                      const float* __restrict__ Win, const float* __restrict__ b_in,
                      const float* __restrict__ Wih_res, const float* __restrict__ Whh_res,
                      const float* __restrict__ bih_res, const float* __restrict__ bhh_res,
                      const float* __restrict__ Wout, const float* __restrict__ bout,
                      float* __restrict__ h_res,
                      const float* __restrict__ Yf, const float* __restrict__ echo,
                      float* __restrict__ out, int t) {
  __shared__ float r0s[B_][MRES_];
  __shared__ float gis[B_][3*MRES_];
  __shared__ float ghs[B_][3*MRES_];
  __shared__ float ths[B_][MRES_];
  int tid = threadIdx.x;
  if (tid < B_*MRES_) {
    int b = tid / MRES_, m = tid % MRES_;
    float s = b_in[m];
    for (int f=0; f<F_; ++f) s += resf[b*F_+f]*Win[m*F_+f];
    r0s[b][m] = tanhf(s);
  }
  __syncthreads();
  for (int k = tid; k < B_*3*MRES_; k += blockDim.x) {
    int b = k / (3*MRES_), j = k % (3*MRES_);
    float gi = bih_res[j], gh = bhh_res[j];
    for (int m=0;m<MRES_;++m){
      gi += r0s[b][m]*Wih_res[j*MRES_+m];
      gh += h_res[b*MRES_+m]*Whh_res[j*MRES_+m];
    }
    gis[b][j]=gi; ghs[b][j]=gh;
  }
  __syncthreads();
  if (tid < B_*MRES_) {
    int b = tid / MRES_, m = tid % MRES_;
    float rg = sigmf_(gis[b][m]         + ghs[b][m]);
    float zg = sigmf_(gis[b][MRES_+m]   + ghs[b][MRES_+m]);
    float ng = tanhf (gis[b][2*MRES_+m] + rg*ghs[b][2*MRES_+m]);
    float hv = (1.f - zg)*ng + zg*h_res[b*MRES_+m];
    h_res[b*MRES_+m] = hv;
    ths[b][m] = tanhf(hv);
  }
  __syncthreads();
  const size_t OFFS = (size_t)BF_*T_*2;
  for (int c = tid; c < BF_; c += blockDim.x) {
    int b = c / F_, f = c % F_;
    float s = bout[f];
    for (int m=0;m<MRES_;++m) s += ths[b][m]*Wout[f*MRES_+m];
    float mask = sigmf_(s);
    float ytr=Yf[((size_t)c*T_+t)*2], yti=Yf[((size_t)c*T_+t)*2+1];
    float dr = ytr - echo[(size_t)c*2], di = yti - echo[(size_t)c*2+1];
    out[((size_t)c*T_+t)*2+0] = mask*dr;           // res_hat
    out[((size_t)c*T_+t)*2+1] = mask*di;
    out[OFFS + ((size_t)c*T_+t)*2+0] = dr;         // s_hat = y - echo
    out[OFFS + ((size_t)c*T_+t)*2+1] = di;
  }
}

extern "C" void kernel_launch(void* const* d_in, const int* in_sizes, int n_in,
                              void* d_out, int out_size, void* d_ws, size_t ws_size,
                              hipStream_t stream) {
  (void)in_sizes; (void)n_in; (void)out_size; (void)ws_size;
  const float* x      = (const float*)d_in[0];
  const float* y      = (const float*)d_in[1];
  const float* Wr1    = (const float*)d_in[2];
  const float* Wi1    = (const float*)d_in[3];
  const float* br1    = (const float*)d_in[4];
  const float* bi1    = (const float*)d_in[5];
  const float* a_in   = (const float*)d_in[6];
  const float* Wih_r  = (const float*)d_in[7];
  const float* Whh_r  = (const float*)d_in[8];
  const float* bih_r  = (const float*)d_in[9];
  const float* bhh_r  = (const float*)d_in[10];
  const float* Wih_i  = (const float*)d_in[11];
  const float* Whh_i  = (const float*)d_in[12];
  const float* bih_i  = (const float*)d_in[13];
  const float* bhh_i  = (const float*)d_in[14];
  const float* Wr2    = (const float*)d_in[15];
  const float* Wi2    = (const float*)d_in[16];
  const float* br2    = (const float*)d_in[17];
  const float* bi2    = (const float*)d_in[18];
  const float* a_out  = (const float*)d_in[19];
  const float* Wr3    = (const float*)d_in[20];
  const float* Wi3    = (const float*)d_in[21];
  const float* br3    = (const float*)d_in[22];
  const float* bi3    = (const float*)d_in[23];
  const float* Win    = (const float*)d_in[24];
  const float* b_in   = (const float*)d_in[25];
  const float* Wih_res= (const float*)d_in[26];
  const float* Whh_res= (const float*)d_in[27];
  const float* bih_res= (const float*)d_in[28];
  const float* bhh_res= (const float*)d_in[29];
  const float* Wout   = (const float*)d_in[30];
  const float* bout   = (const float*)d_in[31];
  float* out = (float*)d_out;

  // ---- workspace carve-up (zeroed region first; fragment pads zeroed once) ----
  char* ws = (char*)d_ws;
  size_t off = 0;
  auto alloc = [&](size_t bytes)->char* {
    char* p = ws + off;
    off = (off + bytes + 255) & ~(size_t)255;
    return p;
  };
  char* zbegin = ws;
  float*     Xfp     = (float*)alloc((size_t)BF_*TP2_*2*4);
  float*     h_prior = (float*)alloc((size_t)BF_*8*4);
  float*     h_post  = (float*)alloc((size_t)BF_*8*4);
  float*     h32     = (float*)alloc((size_t)4*BF_*RD_*4);
  _Float16*  hfrag   = (_Float16*)alloc((size_t)4*FRAGH_*2);
  _Float16*  zfragR  = (_Float16*)alloc((size_t)FRAGH_*2);
  _Float16*  zfragI  = (_Float16*)alloc((size_t)FRAGH_*2);
  _Float16*  ufrag   = (_Float16*)alloc((size_t)FRAGH_*2);
  _Float16*  vfrag   = (_Float16*)alloc((size_t)FRAGH_*2);
  float*     h_res   = (float*)alloc((size_t)B_*MRES_*4);
  size_t zwords = (size_t)((ws + off) - zbegin) / 4;
  float*     Yf      = (float*)alloc((size_t)BF_*T_*2*4);
  float*     e_buf   = (float*)alloc((size_t)BF_*2*4);
  float*     echo    = (float*)alloc((size_t)BF_*2*4);
  float*     resf    = (float*)alloc((size_t)BF_*4);
  _Float16*  wihfrag = (_Float16*)alloc((size_t)2*WFRAGH_*2);
  _Float16*  whhfrag = (_Float16*)alloc((size_t)2*WFRAGH_*2);
  _Float16*  w2frag  = (_Float16*)alloc((size_t)2*W2FRAGH_*2);
  float*     Gi      = (float*)alloc((size_t)4*GSTRIDE_*4);
  float*     Gh      = (float*)alloc((size_t)4*GSTRIDE_*4);
  float*     G2      = (float*)alloc((size_t)2*G2STRIDE_*4);

  // reset recurrent state + fragment pads every call (graph-replay safe)
  k_zero<<<512, 256, 0, stream>>>((uint32_t*)zbegin, zwords);

  // one-time weight swizzles into WMMA B-fragment layout (L2-resident afterwards)
  k_prep_bfrag<<<(WFRAGH_+255)/256, 256, 0, stream>>>(Wih_r, wihfrag,            NG_, FC_, NTG_);
  k_prep_bfrag<<<(WFRAGH_+255)/256, 256, 0, stream>>>(Wih_i, wihfrag + WFRAGH_,  NG_, FC_, NTG_);
  k_prep_bfrag<<<(WFRAGH_+255)/256, 256, 0, stream>>>(Whh_r, whhfrag,            NG_, RD_, NTG_);
  k_prep_bfrag<<<(WFRAGH_+255)/256, 256, 0, stream>>>(Whh_i, whhfrag + WFRAGH_,  NG_, RD_, NTG_);
  k_prep_bfrag<<<(W2FRAGH_+255)/256,256, 0, stream>>>(Wr2,   w2frag,             FC_, RD_, NT2_);
  k_prep_bfrag<<<(W2FRAGH_+255)/256,256, 0, stream>>>(Wi2,   w2frag + W2FRAGH_,  FC_, RD_, NT2_);

  // STFT of both signals
  k_stft<<<2*B_*T_, 256, 0, stream>>>(x, y, Xfp, Yf);

  // sequential Kalman-filter scan
  const int tiles1 = 4*MT_*NTG_;
  const int tiles2 = 2*MT_*NT2_;
  for (int t = 0; t < T_; ++t) {
    k_pre<<<(BF_+255)/256, 256, 0, stream>>>(Xfp, Yf, h_prior, h_post, e_buf,
                                             zfragR, zfragI, Wr1, Wi1, br1, bi1, a_in, t);
    k_gru_gemm<<<(tiles1*32+255)/256, 256, 0, stream>>>(zfragR, zfragI, hfrag,
                                                        wihfrag, whhfrag, Gi, Gh);
    k_gru_combine<<<(BF_*RD_+255)/256, 256, 0, stream>>>(Gi, Gh, bih_r, bhh_r, bih_i, bhh_i,
                                                         h32, hfrag, ufrag, vfrag);
    k_dense2_gemm<<<(tiles2*32+255)/256, 256, 0, stream>>>(ufrag, vfrag, w2frag, G2);
    k_kalman<<<(BF_+255)/256, 256, 0, stream>>>(G2, br2, bi2, a_out, Wr3, Wi3, br3, bi3,
                                                Xfp, Yf, e_buf, h_prior, h_post, echo, resf, t);
    k_res<<<1, 256, 0, stream>>>(resf, Win, b_in, Wih_res, Whh_res, bih_res, bhh_res,
                                 Wout, bout, h_res, Yf, echo, out, t);
  }
}
